// LSTMModel_4406636446292
// MI455X (gfx1250) — matrix operand order repaired and checked by reading, not compiled, not run on hardware
//
#include <hip/hip_runtime.h>
#include <hip/hip_bf16.h>
#include <math.h>

// ---------------------------------------------------------------------------
// LSTMModel on gfx1250: 2-layer LSTM (B=256,T=2048,D=40,H=128) + LN + FC(35)
// Batch-partitioned recurrence: one 16-row batch tile per workgroup, fused
// [x_t | h] @ [W_ih | W_hh]^T step GEMM via v_wmma_f32_16x16x32_bf16.
// Register-resident weights + cell state; x staged via async global->LDS
// (ASYNCcnt) into a double-buffered LDS slot, overlapping barrier+pointwise.
// ---------------------------------------------------------------------------

#define B_   256
#define T_   2048
#define D_   40
#define H_   128
#define C_   35
#define G4H  512          // 4*H
#define KC0  192          // 64 (x padded 40->64) + 128 (h)
#define KC1  256          // 128 (x=hs1)         + 128 (h)

typedef __attribute__((ext_vector_type(16))) __bf16 v16bf;
typedef __attribute__((ext_vector_type(8)))  float  v8f;

struct bf16x16 { uint4 lo, hi; };   // 32 B = 16 bf16 = one WMMA A/B fragment per lane

__device__ __forceinline__ v16bf as_v16bf(bf16x16 p) {
  return __builtin_bit_cast(v16bf, p);
}

__device__ __forceinline__ unsigned short f32_bf16(float f) {
  unsigned int u = __float_as_uint(f);
  u += 0x7FFFu + ((u >> 16) & 1u);          // round-to-nearest-even
  return (unsigned short)(u >> 16);
}

__device__ __forceinline__ float fast_rcp(float x) { return __builtin_amdgcn_rcpf(x); }
__device__ __forceinline__ float fast_sigmoid(float x) {
  return fast_rcp(1.0f + __expf(-x));
}
__device__ __forceinline__ float fast_tanh(float x) {      // branch-free
  float ax = fabsf(x);
  float e  = __expf(-2.0f * ax);
  float t  = (1.0f - e) * fast_rcp(1.0f + e);
  return copysignf(t, x);
}

union F4 { float4 v; float a[4]; };

// async copy of 16 bytes: global -> LDS, tracked by ASYNCcnt
__device__ __forceinline__ void async_b128(unsigned lds_off, const void* gaddr) {
  asm volatile("global_load_async_to_lds_b128 %0, %1, off"
               :: "v"(lds_off), "v"(gaddr) : "memory");
}
__device__ __forceinline__ void wait_async0() {
  asm volatile("s_wait_asynccnt 0" ::: "memory");
}
__device__ __forceinline__ unsigned lds_off32(const void* p) {
  return (unsigned)(size_t)p;    // low 32 bits of shared addr = LDS offset
}

// ---------------------------------------------------------------------------
// Weight prep: bf16 [W_ih | W_hh] row-major [4H][KC] + fused biases.
// ---------------------------------------------------------------------------
__global__ __launch_bounds__(256)
void prep_kernel(const float* __restrict__ Wih0, const float* __restrict__ Whh0,
                 const float* __restrict__ bih0, const float* __restrict__ bhh0,
                 const float* __restrict__ Wih1, const float* __restrict__ Whh1,
                 const float* __restrict__ bih1, const float* __restrict__ bhh1,
                 unsigned short* __restrict__ Wcat0,
                 unsigned short* __restrict__ Wcat1,
                 float* __restrict__ bias0, float* __restrict__ bias1) {
  int i = blockIdx.x * blockDim.x + threadIdx.x;
  if (i < G4H * KC0) {
    int n = i / KC0, k = i % KC0;
    float v = 0.0f;
    if (k < 64) { if (k < D_) v = Wih0[n * D_ + k]; }   // zero-pad K 40..63
    else        { v = Whh0[n * H_ + (k - 64)]; }
    Wcat0[i] = f32_bf16(v);
  }
  if (i < G4H * KC1) {
    int n = i / KC1, k = i % KC1;
    float v = (k < H_) ? Wih1[n * H_ + k] : Whh1[n * H_ + (k - H_)];
    Wcat1[i] = f32_bf16(v);
  }
  if (i < G4H) {
    bias0[i] = bih0[i] + bhh0[i];
    bias1[i] = bih1[i] + bhh1[i];
  }
}

// x: [B][T][D] f32  ->  xbf: [T][B][D] bf16 (time-major for per-step locality)
__global__ __launch_bounds__(256)
void xconv_kernel(const float* __restrict__ x, unsigned short* __restrict__ xbf) {
  size_t i = (size_t)blockIdx.x * blockDim.x + threadIdx.x;
  const size_t total = (size_t)T_ * B_ * D_;
  if (i >= total) return;
  int    d = (int)(i % D_);
  size_t r = i / D_;
  int    b = (int)(r % B_);
  int    t = (int)(r / B_);
  xbf[i] = f32_bf16(x[((size_t)b * T_ + t) * D_ + d]);
}

// ---------------------------------------------------------------------------
// One LSTM layer over all T steps for a 16-row batch tile.
// Block = 512 threads = 16 waves; wave w owns gate columns [w*32, w*32+32).
// Per step: [GEMM + async-stage(t+1)] sync [pointwise] wait_async sync.
// ---------------------------------------------------------------------------
template<int SRCW, bool L0>      // SRCW: source row width (40 or 128 bf16)
__global__ __launch_bounds__(512)
void lstm_layer_kernel(const unsigned short* __restrict__ xsrc, // [T][B][SRCW] bf16
                       const int* __restrict__ length,          // [B]
                       const unsigned short* __restrict__ Wcat, // [4H][KC] bf16
                       const float* __restrict__ bias,          // [4H] fused
                       unsigned short* __restrict__ hs_out,     // L0: [T][B][H] bf16
                       float* __restrict__ hlast) {             // L1: [B][H] f32
  constexpr int XW  = L0 ? 64 : 128;    // padded x width in A matrix
  constexpr int KC  = XW + H_;          // total K
  constexpr int KS  = KC / 32;          // WMMA K steps
  constexpr int XS  = XW / 32;          // K steps from x segment
  constexpr int CPR = SRCW / 8;         // 16B chunks per source row
  constexpr int NST = 16 * CPR;         // staging threads (80 or 256)

  __shared__ alignas(16) unsigned short hbuf[16][H_];     // h (bf16)
  __shared__ alignas(16) unsigned short xbuf[2][16][XW];  // x double buffer
  __shared__ float gates[16][G4H];

  const int tid   = threadIdx.x;
  const int lane  = tid & 31;
  const int wave  = tid >> 5;
  const int m0    = blockIdx.x * 16;         // batch tile base
  const int lhalf = lane >> 4;               // 0: lanes 0-15, 1: lanes 16-31
  const int l15   = lane & 15;

  // ---- pointwise thread ownership: row pm, 4 consecutive cols at pj ----
  const int pm = tid >> 5;                   // 0..15
  const int pj = (tid & 31) << 2;            // 0..124
  const int lenm = length[m0 + pm];
  float hreg[4] = {0.f, 0.f, 0.f, 0.f};
  float creg[4] = {0.f, 0.f, 0.f, 0.f};

  // ---- zero LDS state (h = 0; x pad cols stay 0 forever) ----
  for (int i = tid; i < 2 * 16 * XW; i += 512) ((unsigned short*)xbuf)[i] = 0;
  for (int i = tid; i < 16 * H_;     i += 512) ((unsigned short*)hbuf)[i] = 0;

  // ---- preload B fragments (weights) into registers; reused all 2048 steps.
  // B 32x16 layout: lane holds column n0+(lane&15); two contiguous 8-elem
  // chunks at k0 + 8*lhalf and k0 + 16 + 8*lhalf.
  bf16x16 braw[2][KS];
  #pragma unroll
  for (int nt = 0; nt < 2; ++nt) {
    const int ncol = (wave * 2 + nt) * 16 + l15;
    const unsigned short* wrow = Wcat + (size_t)ncol * KC;
    #pragma unroll
    for (int ks = 0; ks < KS; ++ks) {
      const int koff = ks * 32 + (lhalf << 3);
      braw[nt][ks].lo = *(const uint4*)(wrow + koff);
      braw[nt][ks].hi = *(const uint4*)(wrow + koff + 16);
    }
  }

  // per-lane bias for the two owned output columns (folded into WMMA C init)
  const int cc0 = (wave * 2 + 0) * 16 + l15;
  const int cc1 = (wave * 2 + 1) * 16 + l15;
  const float bv0 = bias[cc0];
  const float bv1 = bias[cc1];

  // ---- async stage: copy step-t source rows straight into xbuf[t&1] ----
  const int se = tid * 8;                    // first element this thread copies
  const int smR = se / SRCW;                 // source row within tile
  const int sdR = se % SRCW;                 // element within row
  const size_t sgbase = (size_t)(m0 + smR) * SRCW + sdR;   // + t*B_*SRCW
  auto stage_async = [&](int t) {
    if (tid < NST) {
      const unsigned short* g = xsrc + sgbase + (size_t)t * B_ * SRCW;
      async_b128(lds_off32(&xbuf[t & 1][smR][sdR]), g);
    }
  };

  __syncthreads();                 // zeros visible before async writes land
  stage_async(0);
  wait_async0();
  __syncthreads();

  for (int t = 0; t < T_; ++t) {
    // ---- step GEMM: gates = [x_t|h] @ Wcat^T + bias  (16 x 512, K=KC) ----
    {
      v8f acc0 = {bv0, bv0, bv0, bv0, bv0, bv0, bv0, bv0};
      v8f acc1 = {bv1, bv1, bv1, bv1, bv1, bv1, bv1, bv1};
      // A 16x32 bf16 layout: lane holds row (lane&15); two contiguous 8-elem
      // chunks at k0 + 8*lhalf and k0 + 16 + 8*lhalf.
      const unsigned short* axrow = &xbuf[t & 1][l15][0];
      const unsigned short* ahrow = &hbuf[l15][0];
      const int kl = lhalf << 3;
      bf16x16 araw[KS];                      // batch all A loads -> few waits
      #pragma unroll
      for (int ks = 0; ks < XS; ++ks) {
        araw[ks].lo = *(const uint4*)(axrow + ks * 32 + kl);
        araw[ks].hi = *(const uint4*)(axrow + ks * 32 + kl + 16);
      }
      #pragma unroll
      for (int ks = XS; ks < KS; ++ks) {
        araw[ks].lo = *(const uint4*)(ahrow + (ks - XS) * 32 + kl);
        araw[ks].hi = *(const uint4*)(ahrow + (ks - XS) * 32 + kl + 16);
      }
      #pragma unroll
      for (int ks = 0; ks < KS; ++ks) {
        const v16bf a = as_v16bf(araw[ks]);
        acc0 = __builtin_amdgcn_wmma_f32_16x16x32_bf16(
            false, a, false, as_v16bf(braw[0][ks]), (short)0, acc0, false, false);
        acc1 = __builtin_amdgcn_wmma_f32_16x16x32_bf16(
            false, a, false, as_v16bf(braw[1][ks]), (short)0, acc1, false, false);
      }
      // C layout: VGPR r -> row r (lanes 0-15) / row 8+r (lanes 16-31)
      const int mrow = lhalf << 3;
      #pragma unroll
      for (int r = 0; r < 8; ++r) {
        gates[mrow + r][cc0] = acc0[r];
        gates[mrow + r][cc1] = acc1[r];
      }
    }

    // async-stage next step's x into the other slot (last read 2 barriers ago)
    if (t + 1 < T_) stage_async(t + 1);

    __syncthreads();   // gates visible; all xbuf/hbuf reads for step t complete

    // ---- pointwise LSTM cell: state fully register-resident ----
    {
      F4 gi, gf, gg, go;
      gi.v = *(const float4*)&gates[pm][pj];
      gf.v = *(const float4*)&gates[pm][H_ + pj];
      gg.v = *(const float4*)&gates[pm][2 * H_ + pj];
      go.v = *(const float4*)&gates[pm][3 * H_ + pj];
      const bool act = (t < lenm);           // freeze state at t >= len
      #pragma unroll
      for (int q = 0; q < 4; ++q) {
        const float i_ = fast_sigmoid(gi.a[q]);
        const float f_ = fast_sigmoid(gf.a[q]);
        const float g_ = fast_tanh(gg.a[q]);
        const float o_ = fast_sigmoid(go.a[q]);
        const float cn = f_ * creg[q] + i_ * g_;
        const float hn = o_ * fast_tanh(cn);
        if (act) { creg[q] = cn; hreg[q] = hn; }
      }
      uint2 hb;
      hb.x = (unsigned)f32_bf16(hreg[0]) | ((unsigned)f32_bf16(hreg[1]) << 16);
      hb.y = (unsigned)f32_bf16(hreg[2]) | ((unsigned)f32_bf16(hreg[3]) << 16);
      *(uint2*)&hbuf[pm][pj] = hb;           // h for next step's A matrix
      if (L0)
        *(uint2*)&hs_out[((size_t)t * B_ + (m0 + pm)) * H_ + pj] = hb;
    }

    // light prefetch two steps ahead (contiguous [T][B][SRCW] block)
    if (t + 2 < T_ && tid < 16)
      __builtin_prefetch(xsrc + ((size_t)(t + 2) * B_ + m0 + tid) * SRCW, 0, 1);

    if (t + 1 < T_) wait_async0();           // staged slot fully in LDS
    __syncthreads();                         // ready for next GEMM
  }

  if (!L0) {
    float4 hv;
    hv.x = hreg[0]; hv.y = hreg[1]; hv.z = hreg[2]; hv.w = hreg[3];
    *(float4*)&hlast[(m0 + pm) * H_ + pj] = hv;
  }
}

// ---------------------------------------------------------------------------
// LayerNorm + FC head: one thread per batch row (trivial cost).
// ---------------------------------------------------------------------------
__global__ __launch_bounds__(128)
void head_kernel(const float* __restrict__ hlast,
                 const float* __restrict__ gamma, const float* __restrict__ beta,
                 const float* __restrict__ fcw,  const float* __restrict__ fcb,
                 float* __restrict__ out) {
  const int b = blockIdx.x * blockDim.x + threadIdx.x;
  if (b >= B_) return;
  const float* h = hlast + b * H_;
  float mu = 0.0f;
  for (int j = 0; j < H_; ++j) mu += h[j];
  mu *= (1.0f / H_);
  float var = 0.0f;
  for (int j = 0; j < H_; ++j) { float d = h[j] - mu; var += d * d; }
  var *= (1.0f / H_);
  const float inv = rsqrtf(var + 1e-5f);
  for (int c = 0; c < C_; ++c) {
    float a = fcb[c];
    const float* w = fcw + c * H_;
    for (int j = 0; j < H_; ++j)
      a += ((h[j] - mu) * inv * gamma[j] + beta[j]) * w[j];
    out[b * C_ + c] = a;
  }
}

// ---------------------------------------------------------------------------
extern "C" void kernel_launch(void* const* d_in, const int* in_sizes, int n_in,
                              void* d_out, int out_size, void* d_ws, size_t ws_size,
                              hipStream_t stream) {
  const float* x     = (const float*)d_in[0];
  const int*   len   = (const int*)  d_in[1];
  const float* Wih0  = (const float*)d_in[2];
  const float* Whh0  = (const float*)d_in[3];
  const float* bih0  = (const float*)d_in[4];
  const float* bhh0  = (const float*)d_in[5];
  const float* Wih1  = (const float*)d_in[6];
  const float* Whh1  = (const float*)d_in[7];
  const float* bih1  = (const float*)d_in[8];
  const float* bhh1  = (const float*)d_in[9];
  const float* ln_g  = (const float*)d_in[10];
  const float* ln_b  = (const float*)d_in[11];
  const float* fcw   = (const float*)d_in[12];
  const float* fcb   = (const float*)d_in[13];
  float* out = (float*)d_out;

  // workspace carve-up (256B aligned)
  char* ws = (char*)d_ws;
  size_t off = 0;
  auto carve = [&](size_t bytes) -> void* {
    void* p = ws + off;
    off = (off + bytes + 255) & ~(size_t)255;
    return p;
  };
  unsigned short* Wcat0 = (unsigned short*)carve((size_t)G4H * KC0 * 2);   // 192 KB
  unsigned short* Wcat1 = (unsigned short*)carve((size_t)G4H * KC1 * 2);   // 256 KB
  float*          bias0 = (float*)carve(G4H * 4);
  float*          bias1 = (float*)carve(G4H * 4);
  float*          hlast = (float*)carve((size_t)B_ * H_ * 4);              // 128 KB
  unsigned short* xbf   = (unsigned short*)carve((size_t)T_ * B_ * D_ * 2);// 40 MB
  unsigned short* hs1   = (unsigned short*)carve((size_t)T_ * B_ * H_ * 2);// 128 MB
  (void)ws_size; (void)in_sizes; (void)n_in; (void)out_size;

  prep_kernel<<<(G4H * KC1 + 255) / 256, 256, 0, stream>>>(
      Wih0, Whh0, bih0, bhh0, Wih1, Whh1, bih1, bhh1, Wcat0, Wcat1, bias0, bias1);

  {
    const size_t total = (size_t)T_ * B_ * D_;
    xconv_kernel<<<(unsigned)((total + 255) / 256), 256, 0, stream>>>(x, xbf);
  }

  lstm_layer_kernel<D_, true><<<B_ / 16, 512, 0, stream>>>(
      xbf, len, Wcat0, bias0, hs1, nullptr);

  lstm_layer_kernel<H_, false><<<B_ / 16, 512, 0, stream>>>(
      hs1, len, Wcat1, bias1, nullptr, hlast);

  head_kernel<<<(B_ + 127) / 128, 128, 0, stream>>>(
      hlast, ln_g, ln_b, fcw, fcb, out);
}
